// VectorQuantizerEMA_79001628443368
// MI455X (gfx1250) — matrix operand order repaired and checked by reading, not compiled, not run on hardware
//
#include <hip/hip_runtime.h>
#include <hip/hip_bf16.h>

#define DIM      256
#define KCODES   8192
#define NROWS    32768
#define BETA_F   0.25f
#define DECAY_F  0.99f
#define EPS_F    1e-05f

#define AS1 __attribute__((address_space(1)))
#define AS3 __attribute__((address_space(3)))

typedef __bf16 bf16;
typedef __attribute__((ext_vector_type(16))) __bf16 v16bf;
typedef __attribute__((ext_vector_type(8)))  __bf16 v8bf;
typedef __attribute__((ext_vector_type(8)))  float  v8f;
typedef __attribute__((ext_vector_type(8)))  float  f32x8;
typedef __attribute__((ext_vector_type(4)))  int    v4i;

union FragAB { v16bf v; v8bf h[2]; bf16 e[16]; };

// ---------------------------------------------------------------------------
// Kernel 1: per-code prep. One block (256 thr) per code row k.
// ---------------------------------------------------------------------------
__global__ __launch_bounds__(256) void vq_init_kernel(
    const float* __restrict__ embedding, const float* __restrict__ ema_cs,
    const float* __restrict__ ema_w,
    bf16* __restrict__ emb_bf, float* __restrict__ enorm,
    float* __restrict__ n_acc, float* __restrict__ cluster_out,
    float* __restrict__ emaw_out, float* __restrict__ loss_out)
{
    __shared__ float ls[8];
    const int k = blockIdx.x;
    const int d = threadIdx.x;
    const size_t off = (size_t)k * DIM + d;
    const float e = embedding[off];
    emb_bf[off]   = (bf16)e;
    emaw_out[off] = DECAY_F * ema_w[off];
    float sq = e * e;
    #pragma unroll
    for (int s = 1; s < 32; s <<= 1) sq += __shfl_xor(sq, s, 32);
    if ((d & 31) == 0) ls[d >> 5] = sq;
    __syncthreads();
    if (d == 0) {
        float t = 0.f;
        #pragma unroll
        for (int w = 0; w < 8; ++w) t += ls[w];
        enorm[k]       = t;
        cluster_out[k] = DECAY_F * ema_cs[k];
    }
    if (k == 0 && d == 0) { *loss_out = 0.0f; *n_acc = 0.0f; }
}

// ---------------------------------------------------------------------------
// Kernel 2: WMMA argmin. One block (4 waves) per 16-row tile of flat z_e.
// score(m,n) = ||e_n||^2 - 2 * <f_m, e_n>.
// Full-tile double-buffered B (two simultaneously-live 64-VGPR buffers):
// tile t+4's 16 global_load_b128 (and its enorm scalar) are in flight while
// tile t's 8 WMMAs execute. enorm is issued BEFORE the B loads of its tile so
// the epilogue dependence resolves with a partial loadcnt wait, not 0x0.
// ---------------------------------------------------------------------------
__global__ __launch_bounds__(128) void vq_argmin_kernel(
    const float* __restrict__ z_e, const bf16* __restrict__ emb_bf,
    const float* __restrict__ enorm, int* __restrict__ indices)
{
    __shared__ __align__(32) float lds_af[16 * DIM];   // 16 KB raw f32 A tile
    __shared__ float red_v[4][16];
    __shared__ int   red_i[4][16];

    const int tid  = threadIdx.x;       // 0..127
    const int wave = tid >> 5;          // 0..3
    const int lane = tid & 31;
    const int row0 = blockIdx.x * 16;

    // ---- Stage the 16x256 f32 A tile into LDS (async DMA if available) ----
    const float* zbase = z_e + (size_t)row0 * DIM;
#if __has_builtin(__builtin_amdgcn_global_load_async_to_lds_b128)
    for (int i = tid; i < (16 * DIM) / 4; i += 128) {
        __builtin_amdgcn_global_load_async_to_lds_b128(
            (AS1 v4i*)(unsigned long long)(zbase + i * 4),
            (AS3 v4i*)(unsigned)(unsigned long long)&lds_af[i * 4],
            0, 0);
    }
#if __has_builtin(__builtin_amdgcn_s_wait_asynccnt)
    __builtin_amdgcn_s_wait_asynccnt(0);
#else
    asm volatile("s_wait_asynccnt 0x0" ::: "memory");
#endif
#else
    for (int i = tid; i < (16 * DIM) / 4; i += 128) {
        ((float4*)lds_af)[i] = ((const float4*)zbase)[i];
    }
#endif
    __syncthreads();

    const int h  = lane >> 4;   // half-wave select
    const int nn = lane & 15;   // row (A) / column (B,C) within tile

    // ---- Build the 8 A fragments (f32 -> bf16 once per block) ----
    // 16-bit A 16x32 striping: elems 0..7 -> K=8h..8h+7, 8..15 -> K=16+8h..
    v16bf afrag[8];
    {
        const float* arow = &lds_af[nn * DIM];
        #pragma unroll
        for (int c = 0; c < 8; ++c) {
            const f32x8 lo = *(const f32x8*)(arow + c * 32 + 8 * h);
            const f32x8 hi = *(const f32x8*)(arow + c * 32 + 16 + 8 * h);
            FragAB fa;
            #pragma unroll
            for (int j = 0; j < 8; ++j) {
                fa.e[j]     = (bf16)lo[j];
                fa.e[8 + j] = (bf16)hi[j];
            }
            afrag[c] = fa.v;
        }
    }

    float bestv[8];
    int   besti[8];
    #pragma unroll
    for (int r = 0; r < 8; ++r) { bestv[r] = INFINITY; besti[r] = 0x7fffffff; }

    // B 32x16 striping: lane column nn, elems 0..7 -> K=8h..8h+7, 8..15 ->
    // K=16+8h.. (matches A); code row slices are contiguous (no transpose).
    auto loadTile = [&](FragAB* buf, float& en, int t) {
        const int code = t * 16 + nn;
        en = enorm[code];                        // issued before the B loads
        const bf16* brow = emb_bf + (size_t)code * DIM + 8 * h;
        #pragma unroll
        for (int c = 0; c < 8; ++c) {
            buf[c].h[0] = *(const v8bf*)(brow + c * 32);
            buf[c].h[1] = *(const v8bf*)(brow + c * 32 + 16);
        }
    };
    auto consume = [&](const FragAB* buf, float en, int t) {
        v8f acc = {};
        #pragma unroll
        for (int c = 0; c < 8; ++c) {
            acc = __builtin_amdgcn_wmma_f32_16x16x32_bf16(
                false, afrag[c], false, buf[c].v, (short)0, acc, false, false);
        }
        const int code = t * 16 + nn;
        // Codes grow monotonically within a wave -> strict '<' keeps the
        // first (lowest-index) minimum; no tie-break needed here.
        #pragma unroll
        for (int r = 0; r < 8; ++r) {
            const float s = fmaf(-2.0f, acc[r], en);
            if (s < bestv[r]) { bestv[r] = s; besti[r] = code; }
        }
    };

    constexpr int NTILES = KCODES / 16;          // 512
    FragAB bufA[8], bufB[8];
    float enA, enB;
    loadTile(bufA, enA, wave);                   // prologue
    for (int t = wave; t < NTILES; t += 8) {     // two tiles per iteration
        loadTile(bufB, enB, t + 4);              // prefetch while consuming A
        if (t + 12 < NTILES)
            __builtin_prefetch(emb_bf + (size_t)((t + 12) * 16 + nn) * DIM, 0, 1);
        consume(bufA, enA, t);
        if (t + 8 < NTILES) loadTile(bufA, enA, t + 8);
        consume(bufB, enB, t + 4);
    }

    // Reduce across the 16 columns: xor shuffles stay inside each 16-lane half.
    #pragma unroll
    for (int r = 0; r < 8; ++r) {
        #pragma unroll
        for (int s = 1; s < 16; s <<= 1) {
            const float ov = __shfl_xor(bestv[r], s, 32);
            const int   oi = __shfl_xor(besti[r], s, 32);
            if (ov < bestv[r] || (ov == bestv[r] && oi < besti[r])) {
                bestv[r] = ov; besti[r] = oi;
            }
        }
    }
    if (nn == 0) {   // lane 0 -> rows 0..7, lane 16 -> rows 8..15
        const int rbase = h * 8;
        #pragma unroll
        for (int r = 0; r < 8; ++r) {
            red_v[wave][rbase + r] = bestv[r];
            red_i[wave][rbase + r] = besti[r];
        }
    }
    __syncthreads();
    if (tid < 16) {
        float bv = red_v[0][tid]; int bi = red_i[0][tid];
        #pragma unroll
        for (int w = 1; w < 4; ++w) {
            const float ov = red_v[w][tid]; const int oi = red_i[w][tid];
            if (ov < bv || (ov == bv && oi < bi)) { bv = ov; bi = oi; }
        }
        indices[row0 + tid] = bi;
    }
}

// ---------------------------------------------------------------------------
// Kernel 3: gather z_q, loss, and EMA scatter. One block (256 thr) per row.
// ---------------------------------------------------------------------------
__global__ __launch_bounds__(256) void vq_gather_scatter_kernel(
    const float* __restrict__ z_e, const float* __restrict__ embedding,
    const int* __restrict__ indices,
    float* __restrict__ zq_out, float* __restrict__ loss_out,
    float* __restrict__ cluster_out, float* __restrict__ emaw_out)
{
    __shared__ float ls[8];
    const int row = blockIdx.x;
    const int d   = threadIdx.x;
    const int idx = indices[row];
    const float e = embedding[(size_t)idx * DIM + d];
    const float z = z_e[(size_t)row * DIM + d];
    zq_out[(size_t)row * DIM + d] = e;            // z_e + sg(z_q - z_e) == z_q
    atomicAdd(&emaw_out[(size_t)idx * DIM + d], (1.0f - DECAY_F) * z);
    const float diff = e - z;
    float sq = diff * diff;
    #pragma unroll
    for (int s = 1; s < 32; s <<= 1) sq += __shfl_xor(sq, s, 32);
    if ((d & 31) == 0) ls[d >> 5] = sq;
    __syncthreads();
    if (d == 0) {
        float t = 0.f;
        #pragma unroll
        for (int w = 0; w < 8; ++w) t += ls[w];
        const float scale = (1.0f + BETA_F) / ((float)NROWS * (float)DIM);
        atomicAdd(loss_out, t * scale);
        atomicAdd(&cluster_out[idx], 1.0f - DECAY_F);
    }
}

// ---------------------------------------------------------------------------
// Kernel 4: n = sum(new_cluster). 32 blocks x 256.
// ---------------------------------------------------------------------------
__global__ __launch_bounds__(256) void vq_cluster_sum_kernel(
    const float* __restrict__ cluster, float* __restrict__ n_acc)
{
    __shared__ float ls[8];
    const int i = blockIdx.x * 256 + threadIdx.x;
    float v = cluster[i];
    #pragma unroll
    for (int s = 1; s < 32; s <<= 1) v += __shfl_xor(v, s, 32);
    if ((threadIdx.x & 31) == 0) ls[threadIdx.x >> 5] = v;
    __syncthreads();
    if (threadIdx.x == 0) {
        float t = 0.f;
        #pragma unroll
        for (int w = 0; w < 8; ++w) t += ls[w];
        atomicAdd(n_acc, t);
    }
}

// ---------------------------------------------------------------------------
// Kernel 5: new_embedding = new_ema_w / ((new_cluster+eps)/(n+K*eps)*n)
// ---------------------------------------------------------------------------
__global__ __launch_bounds__(256) void vq_finalize_kernel(
    const float* __restrict__ emaw_out, const float* __restrict__ cluster_out,
    const float* __restrict__ n_acc, float* __restrict__ emb_out)
{
    const int k = blockIdx.x;
    const int d = threadIdx.x;
    const float n  = *n_acc;
    const float cs = (cluster_out[k] + EPS_F) / (n + (float)KCODES * EPS_F) * n;
    emb_out[(size_t)k * DIM + d] = emaw_out[(size_t)k * DIM + d] / cs;
}

// ---------------------------------------------------------------------------
extern "C" void kernel_launch(void* const* d_in, const int* in_sizes, int n_in,
                              void* d_out, int out_size, void* d_ws, size_t ws_size,
                              hipStream_t stream) {
    (void)in_sizes; (void)n_in; (void)out_size; (void)ws_size;
    const float* z_e       = (const float*)d_in[0];   // [32768, 256]
    const float* embedding = (const float*)d_in[1];   // [8192, 256]
    const float* ema_cs    = (const float*)d_in[2];   // [8192]
    const float* ema_w     = (const float*)d_in[3];   // [8192, 256]

    // Output layout (flat concat of the reference's return tuple):
    float* out = (float*)d_out;
    const size_t OFF_LOSS    = (size_t)NROWS * DIM;           // 8388608
    const size_t OFF_EMB     = OFF_LOSS + 1;                  // 8388609
    const size_t OFF_CLUSTER = OFF_EMB + (size_t)KCODES*DIM;  // 10485761
    const size_t OFF_EMAW    = OFF_CLUSTER + KCODES;          // 10493953
    float* zq_out      = out;
    float* loss_out    = out + OFF_LOSS;
    float* emb_out     = out + OFF_EMB;
    float* cluster_out = out + OFF_CLUSTER;
    float* emaw_out    = out + OFF_EMAW;

    // Workspace layout (~4.4 MB).
    char* ws = (char*)d_ws;
    bf16*  emb_bf  = (bf16*)ws;                                  // K*D*2
    float* enorm   = (float*)(ws + (size_t)KCODES * DIM * 2);    // K*4
    int*   indices = (int*)(ws + (size_t)KCODES * DIM * 2 + (size_t)KCODES * 4);
    float* n_acc   = (float*)(ws + (size_t)KCODES * DIM * 2 + (size_t)KCODES * 4
                                 + (size_t)NROWS * 4);

    vq_init_kernel<<<KCODES, 256, 0, stream>>>(
        embedding, ema_cs, ema_w, emb_bf, enorm, n_acc,
        cluster_out, emaw_out, loss_out);

    vq_argmin_kernel<<<NROWS / 16, 128, 0, stream>>>(
        z_e, emb_bf, enorm, indices);

    vq_gather_scatter_kernel<<<NROWS, 256, 0, stream>>>(
        z_e, embedding, indices, zq_out, loss_out, cluster_out, emaw_out);

    vq_cluster_sum_kernel<<<KCODES / 256, 256, 0, stream>>>(cluster_out, n_acc);

    vq_finalize_kernel<<<KCODES, 256, 0, stream>>>(
        emaw_out, cluster_out, n_acc, emb_out);
}
